// GNN_new_13623636263487
// MI455X (gfx1250) — matrix operand order repaired and checked by reading, not compiled, not run on hardware
//
#include <hip/hip_runtime.h>
#include <math.h>

typedef __bf16 bf16_t;
typedef __attribute__((ext_vector_type(16))) __bf16 v16bf;
typedef __attribute__((ext_vector_type(8)))  __bf16 v8bf;
typedef __attribute__((ext_vector_type(8)))  float  v8f;

#define DDIM   512
#define NNODE  256
#define NBATCH 64
#define NROWS  (NBATCH * NNODE)   /* 16384 */
#define NJOINT (5 * DDIM)         /* 2560  */

// ---- CDNA5 async global->LDS path (guarded; fallback = regular load+store) ----
#if defined(__has_builtin)
# if __has_builtin(__builtin_amdgcn_global_load_async_to_lds_b128)
#  define HAVE_ASYNC_LDS 1
# endif
#endif
#ifndef HAVE_ASYNC_LDS
# define HAVE_ASYNC_LDS 0
#endif

// Builtin signature (from the compiler diagnostic): param 1 is a NON-const
// global-AS pointer to a GCC-vector int4; match it exactly.
typedef int v4i_gcc __attribute__((vector_size(16)));
#define GPTR(p) ((__attribute__((address_space(1))) v4i_gcc*)(p))
#define LPTR(p) ((__attribute__((address_space(3))) v4i_gcc*)(p))

__device__ __forceinline__ void async_wait_and_barrier() {
#if HAVE_ASYNC_LDS
  asm volatile("s_wait_asynccnt 0x0" ::: "memory");
#endif
  __syncthreads();
}

__device__ __forceinline__ void copy16(uint4* dst, const char* src) {
#if HAVE_ASYNC_LDS
  __builtin_amdgcn_global_load_async_to_lds_b128(GPTR(src), LPTR(dst), 0, 0);
#else
  *dst = *(const uint4*)src;
#endif
}

__device__ __forceinline__ v8f vzero8() {
  v8f z;
#pragma unroll
  for (int i = 0; i < 8; ++i) z[i] = 0.0f;
  return z;
}

// A-fragment (16x32 bf16, MxK) from a row-major bf16 matrix.
// Contiguous k order used here (B pack matches): lanes 0-15 -> M=lane, K in {0..15}; lanes 16-31 -> K in {16..31}.
__device__ __forceinline__ v16bf load_a_frag(const bf16_t* __restrict__ base, int ld,
                                             int rowbase, int kt, int lane) {
  int m  = rowbase + (lane & 15);
  int kh = kt * 32 + ((lane & 16) ? 16 : 0);
  const bf16_t* p = base + m * ld + kh;
  v8bf lo = *(const v8bf*)(p);
  v8bf hi = *(const v8bf*)(p + 8);
  v16bf a;
#pragma unroll
  for (int i = 0; i < 8; ++i) { a[i] = lo[i]; a[i + 8] = hi[i]; }
  return a;
}

// B-fragment from an LDS slab of packed tiles: tile t, 32 contiguous bytes per lane.
__device__ __forceinline__ v16bf lds_b_frag(const bf16_t* lbase, int t, int lane) {
  return *(const v16bf*)(lbase + t * 256 + lane * 16);
}

__device__ __forceinline__ v8f wmma_bf16(v16bf a, v16bf b, v8f c) {
  return __builtin_amdgcn_wmma_f32_16x16x32_bf16(false, a, false, b, (short)0, c, false, false);
}

// ---------------- conversion / packing kernels ----------------

__global__ void conv_mask_kernel(const float* __restrict__ src, bf16_t* __restrict__ dst, int n) {
  int i = blockIdx.x * blockDim.x + threadIdx.x;
  if (i < n) dst[i] = (bf16_t)src[i];
}

// h: fp32 (NROWS x DDIM) -> bf16 row-major + bf16 B-packed (per-batch 256x512 B matrices).
__global__ void conv_h_kernel(const float* __restrict__ h, bf16_t* __restrict__ hrow,
                              bf16_t* __restrict__ hpack) {
  int i = blockIdx.x * blockDim.x + threadIdx.x;  // exactly NROWS*DDIM threads
  float f = h[i];
  bf16_t v = (bf16_t)f;
  hrow[i] = v;
  int g = i >> 9;          // global row
  int n = i & 511;         // column
  int b = g >> 8, k = g & 255;
  int kt = k >> 5, kr = k & 31;
  int off = ((((b * 8 + kt) * 32 + (n >> 4)) * 32) + ((n & 15) | (kr & 16))) * 16 + (kr & 15);
  hpack[off] = v;
}

// Weight (K x N, row-major fp32) -> bf16 B-fragment pack.
__global__ void pack_b_kernel(const float* __restrict__ W, bf16_t* __restrict__ out, int K, int N) {
  int i = blockIdx.x * blockDim.x + threadIdx.x;
  if (i >= K * N) return;
  int k = i / N, n = i - k * N;
  int NT = N >> 4;
  int kt = k >> 5, kr = k & 31;
  int off = ((kt * NT + (n >> 4)) * 32 + ((n & 15) | (kr & 16))) * 16 + (kr & 15);
  out[off] = (bf16_t)W[i];
}

// ---------------- GEMM kernels ----------------
// Common block shape: 256 threads = 8 waves. All 8 waves share the same B column
// group; B tiles are staged (async) into a double-buffered LDS slab once per
// k-step and consumed by all waves -> 8x less L2 traffic, loads overlap WMMA.

// K1: agg[b] = mask[b] (256x256) @ h[b] (256x512).
// grid = 1024 blocks: b(64) x cg(8) x rtg(2); wave -> rt = rtg*8 + wslot.
__global__ void k1_mask_gemm(const bf16_t* __restrict__ maskb, const bf16_t* __restrict__ hpack,
                             bf16_t* __restrict__ agg) {
  __shared__ uint4 smem[256];  // 2 slabs x 128 chunks (2 KB each: 4 B-tiles)
  int tid = threadIdx.x, lane = tid & 31, wslot = tid >> 5;
  int rtg = blockIdx.x & 1, cg = (blockIdx.x >> 1) & 7, b = blockIdx.x >> 4;
  int rt = rtg * 8 + wslot;
  const bf16_t* A = maskb + b * 256 * 256;
  const char* bsrc = (const char*)hpack;
  v8f c[4];
#pragma unroll
  for (int t = 0; t < 4; ++t) c[t] = vzero8();
  // stage kt=0
  if (tid < 128)
    copy16(smem + tid, bsrc + (((size_t)((b * 8 + 0) * 32 + cg * 4)) << 9) + tid * 16);
  for (int kt = 0; kt < 8; ++kt) {
    int buf = kt & 1;
    async_wait_and_barrier();
    if (kt + 1 < 8 && tid < 128)
      copy16(smem + (buf ^ 1) * 128 + tid,
             bsrc + (((size_t)((b * 8 + kt + 1) * 32 + cg * 4)) << 9) + tid * 16);
    v16bf a = load_a_frag(A, 256, rt * 16, kt, lane);
    const bf16_t* lb = (const bf16_t*)(smem + buf * 128);
    v16bf bf[4];
#pragma unroll
    for (int t = 0; t < 4; ++t) bf[t] = lds_b_frag(lb, t, lane);
#pragma unroll
    for (int t = 0; t < 4; ++t) c[t] = wmma_bf16(a, bf[t], c[t]);
  }
  int mrow = rt * 16 + ((lane & 16) ? 8 : 0);
  int ncol = cg * 64 + (lane & 15);
#pragma unroll
  for (int t = 0; t < 4; ++t)
#pragma unroll
    for (int v = 0; v < 8; ++v)
      agg[(b * 256 + mrow + v) * DDIM + ncol + t * 16] = (bf16_t)c[t][v];
}

// K2: nb = agg (16384x512) @ W_p (512x512) + 256*b_p.
// grid = 1024 blocks: rtg(128) x cg(8); wave -> rt = rtg*8 + wslot.
__global__ void k2_wp_gemm(const bf16_t* __restrict__ aggb, const bf16_t* __restrict__ wpack,
                           const float* __restrict__ b_p, bf16_t* __restrict__ nb) {
  __shared__ uint4 smem[256];
  int tid = threadIdx.x, lane = tid & 31, wslot = tid >> 5;
  int cg = blockIdx.x & 7, rtg = blockIdx.x >> 3;
  int rt = rtg * 8 + wslot;
  const char* bsrc = (const char*)wpack;
  v8f c[4];
#pragma unroll
  for (int t = 0; t < 4; ++t) c[t] = vzero8();
  if (tid < 128)
    copy16(smem + tid, bsrc + (((size_t)(0 * 32 + cg * 4)) << 9) + tid * 16);
  for (int kt = 0; kt < 16; ++kt) {
    int buf = kt & 1;
    async_wait_and_barrier();
    if (kt + 1 < 16 && tid < 128)
      copy16(smem + (buf ^ 1) * 128 + tid,
             bsrc + (((size_t)((kt + 1) * 32 + cg * 4)) << 9) + tid * 16);
    v16bf a = load_a_frag(aggb, DDIM, rt * 16, kt, lane);
    const bf16_t* lb = (const bf16_t*)(smem + buf * 128);
    v16bf bf[4];
#pragma unroll
    for (int t = 0; t < 4; ++t) bf[t] = lds_b_frag(lb, t, lane);
#pragma unroll
    for (int t = 0; t < 4; ++t) c[t] = wmma_bf16(a, bf[t], c[t]);
  }
  int mrow = rt * 16 + ((lane & 16) ? 8 : 0);
  int ncol = cg * 64 + (lane & 15);
#pragma unroll
  for (int t = 0; t < 4; ++t) {
    float bias = 256.0f * b_p[ncol + t * 16];
#pragma unroll
    for (int v = 0; v < 8; ++v)
      nb[(mrow + v) * DDIM + ncol + t * 16] = (bf16_t)(c[t][v] + bias);
  }
}

// K3: fused dual joint GEMM + bias + product + group-of-5 pool + signed sqrt -> iq (d_out).
// grid = 4096 blocks: rtg(128) x cg(32); wave -> rt = rtg*8 + wslot; 16row x 80col tile.
__global__ void k3_joint(const bf16_t* __restrict__ hrow, const bf16_t* __restrict__ nbrow,
                         const bf16_t* __restrict__ wnode, const bf16_t* __restrict__ wnb,
                         const float* __restrict__ b_node, const float* __restrict__ b_nb,
                         float* __restrict__ iq) {
  __shared__ uint4 smem[640];      // 2 slabs x 320 chunks (5 KB each: 5+5 B-tiles)
  __shared__ float pool[8 * 1280]; // 16x80 fp32 per wave
  int tid = threadIdx.x, lane = tid & 31, wslot = tid >> 5;
  int cg = blockIdx.x & 31, rtg = blockIdx.x >> 5;
  int rt = rtg * 8 + wslot;
  int rowbase = rt * 16;
  const char* src_n = (const char*)wnode;
  const char* src_b = (const char*)wnb;
  const int NT = NJOINT >> 4;  // 160
  v8f d[5], e[5];
#pragma unroll
  for (int t = 0; t < 5; ++t) { d[t] = vzero8(); e[t] = vzero8(); }

  // stage(kt, buf): 320 x 16B chunks; [0,160) from wnode, [160,320) from wnb
  {
    size_t base = ((size_t)(0 * NT + cg * 5)) << 9;
    const char* s = (tid < 160) ? (src_n + base + tid * 16) : (src_b + base + (tid - 160) * 16);
    copy16(smem + tid, s);
    if (tid < 64) copy16(smem + 256 + tid, src_b + base + (96 + tid) * 16);
  }
  for (int kt = 0; kt < 16; ++kt) {
    int buf = kt & 1;
    async_wait_and_barrier();
    if (kt + 1 < 16) {
      size_t base = ((size_t)((kt + 1) * NT + cg * 5)) << 9;
      uint4* dst = smem + (buf ^ 1) * 320;
      const char* s = (tid < 160) ? (src_n + base + tid * 16) : (src_b + base + (tid - 160) * 16);
      copy16(dst + tid, s);
      if (tid < 64) copy16(dst + 256 + tid, src_b + base + (96 + tid) * 16);
    }
    v16bf ah = load_a_frag(hrow, DDIM, rowbase, kt, lane);
    v16bf an = load_a_frag(nbrow, DDIM, rowbase, kt, lane);
    const bf16_t* lb = (const bf16_t*)(smem + buf * 320);
    v16bf bn[5], bb[5];
#pragma unroll
    for (int t = 0; t < 5; ++t) {
      bn[t] = lds_b_frag(lb, t, lane);
      bb[t] = lds_b_frag(lb + 1280, t, lane);
    }
#pragma unroll
    for (int t = 0; t < 5; ++t) {
      d[t] = wmma_bf16(ah, bn[t], d[t]);
      e[t] = wmma_bf16(an, bb[t], e[t]);
    }
  }
  // epilogue: bias + product -> wave-private LDS slab, then group-of-5 pool
  float* myl = pool + wslot * 1280;
  int mloc = (lane & 16) ? 8 : 0;
  int nl = lane & 15;
#pragma unroll
  for (int t = 0; t < 5; ++t) {
    int col = cg * 80 + t * 16 + nl;
    float bd = b_node[col], be = b_nb[col];
#pragma unroll
    for (int v = 0; v < 8; ++v)
      myl[(mloc + v) * 80 + t * 16 + nl] = (d[t][v] + bd) * (e[t][v] + be);
  }
  __syncthreads();
#pragma unroll
  for (int r = 0; r < 8; ++r) {
    int idx = r * 32 + lane;
    int m = idx >> 4, p = idx & 15;
    float s = 0.0f;
#pragma unroll
    for (int q = 0; q < 5; ++q) s += myl[m * 80 + p * 5 + q];
    iq[(rt * 16 + m) * DDIM + cg * 16 + p] = copysignf(sqrtf(fabsf(s)), s);
  }
}

// K4: per-row L2 normalize (in-place fp32) + emit bf16 row-major and bf16 B-packed h.
__global__ void k4_norm(float* __restrict__ io, bf16_t* __restrict__ hrow,
                        bf16_t* __restrict__ hpack) {
  __shared__ float wsum[2];
  int row = blockIdx.x, tid = threadIdx.x;  // 64 threads, 2 waves
  const float4* p = (const float4*)(io + row * DDIM);
  float4 x0 = p[tid * 2];
  float4 x1 = p[tid * 2 + 1];
  float vals[8] = {x0.x, x0.y, x0.z, x0.w, x1.x, x1.y, x1.z, x1.w};
  float s = 0.0f;
#pragma unroll
  for (int i = 0; i < 8; ++i) s += vals[i] * vals[i];
#pragma unroll
  for (int off = 16; off > 0; off >>= 1) s += __shfl_xor(s, off, 32);
  if ((tid & 31) == 0) wsum[tid >> 5] = s;
  __syncthreads();
  float inv = 1.0f / fmaxf(sqrtf(wsum[0] + wsum[1]), 1e-12f);
  int b = row >> 8, k = row & 255;
  int kt = k >> 5, kr = k & 31;
  int lanebit = kr & 16, kk = kr & 15;
#pragma unroll
  for (int i = 0; i < 8; ++i) {
    int n = tid * 8 + i;
    float h = vals[i] * inv;
    io[row * DDIM + n] = h;
    bf16_t v = (bf16_t)h;
    hrow[row * DDIM + n] = v;
    int off = ((((b * 8 + kt) * 32 + (n >> 4)) * 32) + ((n & 15) | lanebit)) * 16 + kk;
    hpack[off] = v;
  }
}

// ---------------- launch ----------------

extern "C" void kernel_launch(void* const* d_in, const int* in_sizes, int n_in,
                              void* d_out, int out_size, void* d_ws, size_t ws_size,
                              hipStream_t stream) {
  (void)in_sizes; (void)n_in; (void)out_size; (void)ws_size;
  const float* cn     = (const float*)d_in[0];
  const float* mask   = (const float*)d_in[1];
  const float* W_p    = (const float*)d_in[2];
  const float* b_p    = (const float*)d_in[3];
  const float* W_node = (const float*)d_in[4];
  const float* b_node = (const float*)d_in[5];
  const float* W_nb   = (const float*)d_in[6];
  const float* b_nb   = (const float*)d_in[7];
  float* out = (float*)d_out;

  char* ws = (char*)d_ws;
  const size_t MB = 1u << 20;
  bf16_t* h_row    = (bf16_t*)(ws + 0 * MB);    // 16 MB
  bf16_t* h_pack   = (bf16_t*)(ws + 16 * MB);   // 16 MB
  bf16_t* mask_b   = (bf16_t*)(ws + 32 * MB);   // 8 MB
  bf16_t* agg_b    = (bf16_t*)(ws + 40 * MB);   // 16 MB
  bf16_t* nb_b     = (bf16_t*)(ws + 56 * MB);   // 16 MB
  bf16_t* wp_pk    = (bf16_t*)(ws + 72 * MB);   // 0.5 MB
  bf16_t* wnode_pk = (bf16_t*)(ws + 73 * MB);   // 2.5 MB
  bf16_t* wnb_pk   = (bf16_t*)(ws + 76 * MB);   // 2.5 MB  (total 79 MB)

  int nmask = NBATCH * NNODE * NNODE;  // 4,194,304
  conv_mask_kernel<<<nmask / 256, 256, 0, stream>>>(mask, mask_b, nmask);
  conv_h_kernel<<<(NROWS * DDIM) / 256, 256, 0, stream>>>(cn, h_row, h_pack);
  pack_b_kernel<<<(DDIM * DDIM + 255) / 256, 256, 0, stream>>>(W_p, wp_pk, DDIM, DDIM);
  pack_b_kernel<<<(DDIM * NJOINT + 255) / 256, 256, 0, stream>>>(W_node, wnode_pk, DDIM, NJOINT);
  pack_b_kernel<<<(DDIM * NJOINT + 255) / 256, 256, 0, stream>>>(W_nb, wnb_pk, DDIM, NJOINT);

  for (int step = 0; step < 3; ++step) {
    k1_mask_gemm<<<1024, 256, 0, stream>>>(mask_b, h_pack, agg_b);
    k2_wp_gemm<<<1024, 256, 0, stream>>>(agg_b, wp_pk, b_p, nb_b);
    k3_joint<<<4096, 256, 0, stream>>>(h_row, nb_b, wnode_pk, wnb_pk, b_node, b_nb, out);
    k4_norm<<<NROWS, 64, 0, stream>>>(out, h_row, h_pack);
  }
}
